// Layer_25400436588628
// MI455X (gfx1250) — matrix operand order repaired and checked by reading, not compiled, not run on hardware
//
#include <hip/hip_runtime.h>
#include <stddef.h>

typedef _Float16 f16_t;
typedef __attribute__((ext_vector_type(16))) _Float16 v16h;
typedef __attribute__((ext_vector_type(4)))  _Float16 v4h;
typedef __attribute__((ext_vector_type(2)))  _Float16 v2h;
typedef __attribute__((ext_vector_type(8)))  float    v8f;
typedef __attribute__((ext_vector_type(4)))  unsigned int u32x4;

namespace {
constexpr int kM = 8192;
constexpr int kN = 8192;
constexpr int kK = 4096;
constexpr int BM = 256;           // block tile M
constexpr int BN = 128;           // block tile N
constexpr int BK = 32;            // one f16-WMMA K step per tile
constexpr int LDA = BK + 8;       // 40 halves = 80B pitch (conflict-free 4-dword spacing)
constexpr int LDB = BK + 8;
constexpr int KTILES = kK / BK;   // 128
}

__device__ __forceinline__ v8f wmma_f16(v16h a, v16h b, v8f c) {
  // D = A(16x32 f16) * B(32x16 f16) + C(16x16 f32)
  return __builtin_amdgcn_wmma_f32_16x16x32_f16(
      /*neg_a=*/false, a, /*neg_b=*/false, b,
      /*c_mod=*/(short)0, c, /*reuse_a=*/false, /*reuse_b=*/false);
}

__global__ __launch_bounds__(256)
void w4a16_wmma_gemm(const float* __restrict__ A, const int* __restrict__ Bp,
                     const float* __restrict__ S, float* __restrict__ C) {
  __shared__ alignas(16) f16_t Als[2][BM * LDA];  // row-major, k contiguous (20KB/buf)
  __shared__ alignas(16) f16_t Bls[2][BN * LDB];  // column-major, k contiguous (10KB/buf)

  const int tid  = threadIdx.x;
  const int lane = tid & 31;
  const int wave = tid >> 5;
  const int lh   = lane & 15;   // fragment row/col within 16
  const int sel  = lane >> 4;   // which half of the wave

  const int m_blk = blockIdx.y * BM;
  const int n_blk = blockIdx.x * BN;

  // 8 waves: 4 along M x 2 along N, each wave owns a 64x64 sub-tile
  const int wm = (wave & 3) * 64;
  const int wn = (wave >> 2) * 64;

  v8f acc[4][4];                 // 16 independent WMMA accumulator chains
  const v8f zero8 = {0.f, 0.f, 0.f, 0.f, 0.f, 0.f, 0.f, 0.f};
#pragma unroll
  for (int i = 0; i < 4; ++i)
#pragma unroll
    for (int j = 0; j < 4; ++j) acc[i][j] = zero8;

  // ---- global staging registers ----
  float4 apre[8];   // 256x32 f32 tile: 2048 float4 / 256 threads = 8 each
  int    bpre[2];   // 4 packed-k rows x 128 cols: 512 i32 / 256 threads = 2 each
  float  spre;

  const int bn_col = tid & 127;   // B column this thread dequantizes
  const int bkp0   = tid >> 7;    // first packed-k row (0/1); second at +2

  auto load_globals = [&](int kt) {
    const int kb = kt * BK;
#pragma unroll
    for (int i = 0; i < 8; ++i) {
      const int f   = tid + i * 256;   // 0..2047
      const int row = f >> 3;          // 8 float4 per 32-float row
      const int c4  = f & 7;
      apre[i] = *(const float4*)(A + (size_t)(m_blk + row) * kK + kb + c4 * 4);
    }
    spre = S[(size_t)(kb >> 7) * kN + n_blk + bn_col];   // group = kb/128
#pragma unroll
    for (int i = 0; i < 2; ++i) {
      const int kpr = bkp0 + i * 2;    // 0..3 packed rows (8 k each)
      bpre[i] = Bp[(size_t)((kb >> 3) + kpr) * kN + n_blk + bn_col];
    }
  };

  auto store_lds = [&](int buf) {
#pragma unroll
    for (int i = 0; i < 8; ++i) {
      const int f   = tid + i * 256;
      const int row = f >> 3;
      const int c4  = f & 7;
      v4h h;
      h[0] = (f16_t)apre[i].x; h[1] = (f16_t)apre[i].y;
      h[2] = (f16_t)apre[i].z; h[3] = (f16_t)apre[i].w;
      *(v4h*)&Als[buf][row * LDA + c4 * 4] = h;   // ds_store_b64
    }
    // ---- int4 -> f16 dequant, f16 magic-bias (exact) ----
    // bits 0x6400|q == (f16)(1024+q);  (1024+q) - 1032 == q-8 exactly in f16.
    v2h hs2; hs2[0] = (f16_t)spre; hs2[1] = hs2[0];
    union U2 { unsigned u; v2h h; };
    U2 c1032; c1032.u = 0x64086408u;   // {1032.0h, 1032.0h}
#pragma unroll
    for (int i = 0; i < 2; ++i) {
      const int kpr = bkp0 + i * 2;
      const unsigned w = (unsigned)bpre[i];
      union { v2h h2[4]; u32x4 q; } pk;
#pragma unroll
      for (int j = 0; j < 4; ++j) {             // nibbles (2j, 2j+1)
        const unsigned t = w >> (8 * j);
        U2 qb;
        qb.u = (t & 0xFu) | ((t << 12) & 0xF0000u) | 0x64006400u;
        pk.h2[j] = (qb.h - c1032.h) * hs2;      // v_pk_add_f16 + v_pk_mul_f16
      }
      *(u32x4*)&Bls[buf][bn_col * LDB + kpr * 8] = pk.q;   // ds_store_b128
    }
  };

  // A fragment (16x32): lane<16 holds K{0..7,16..23}, lane>=16 K{8..15,24..31}
  auto load_a_frag = [&](int buf, int mi) -> v16h {
    const f16_t* p = &Als[buf][(wm + mi * 16 + lh) * LDA];
    union { u32x4 q[2]; v16h h; } f;
    f.q[0] = *(const u32x4*)(p + sel * 8);        // ds_load_b128
    f.q[1] = *(const u32x4*)(p + 16 + sel * 8);   // ds_load_b128
    return f.h;
  };
  // B fragment (32x16): lane = column, lane<16 K{0..7,16..23}, lane>=16 K{8..15,24..31}
  auto load_b_frag = [&](int buf, int nj) -> v16h {
    const f16_t* p = &Bls[buf][(wn + nj * 16 + lh) * LDB];
    union { u32x4 q[2]; v16h h; } f;
    f.q[0] = *(const u32x4*)(p + sel * 8);
    f.q[1] = *(const u32x4*)(p + 16 + sel * 8);
    return f.h;
  };

  // ---- pipeline prologue ----
  load_globals(0);
  store_lds(0);
  __syncthreads();

  for (int kt = 0; kt < KTILES; ++kt) {
    const int cur = kt & 1;
    if (kt + 1 < KTILES) load_globals(kt + 1);   // overlap with compute below

    // 16 ds_load_b128 feed 16 independent WMMAs (1.0 loads/wmma)
    v16h af[4], bf[4];
#pragma unroll
    for (int mi = 0; mi < 4; ++mi) af[mi] = load_a_frag(cur, mi);
#pragma unroll
    for (int nj = 0; nj < 4; ++nj) bf[nj] = load_b_frag(cur, nj);
#pragma unroll
    for (int nj = 0; nj < 4; ++nj)
#pragma unroll
      for (int mi = 0; mi < 4; ++mi)
        acc[mi][nj] = wmma_f16(af[mi], bf[nj], acc[mi][nj]);

    __syncthreads();                 // all reads of lds[cur^1] (prev iter) done
    if (kt + 1 < KTILES) store_lds(cur ^ 1);
    __syncthreads();                 // new tile visible to all waves
  }

  // ---- epilogue: C fragment layout: VGPR r -> M = r + sel*8, N = lh ----
  // Non-temporal: C is written once and never re-read; keep L2 for A/B.
#pragma unroll
  for (int mi = 0; mi < 4; ++mi) {
#pragma unroll
    for (int nj = 0; nj < 4; ++nj) {
      const int col  = n_blk + wn + nj * 16 + lh;
      const int row0 = m_blk + wm + mi * 16 + sel * 8;
      float* cp = C + (size_t)row0 * kN + col;
#pragma unroll
      for (int r = 0; r < 8; ++r) {
        __builtin_nontemporal_store(acc[mi][nj][r], cp + (size_t)r * kN);
      }
    }
  }
}

extern "C" void kernel_launch(void* const* d_in, const int* in_sizes, int n_in,
                              void* d_out, int out_size, void* d_ws, size_t ws_size,
                              hipStream_t stream) {
  (void)in_sizes; (void)n_in; (void)out_size; (void)d_ws; (void)ws_size;
  const float* A  = (const float*)d_in[0];   // (M, K) fp32
  const int*   Bp = (const int*)d_in[1];     // (K/8, N) int32 (8 nibbles along k)
  const float* S  = (const float*)d_in[2];   // (K/128, N) fp32
  float* C = (float*)d_out;                  // (M, N) fp32

  dim3 grid(kN / BN, kM / BM);
  w4a16_wmma_gemm<<<grid, 256, 0, stream>>>(A, Bp, S, C);
}
